// KMeansClustering_86784109183355
// MI455X (gfx1250) — matrix operand order repaired and checked by reading, not compiled, not run on hardware
//
#include <hip/hip_runtime.h>
#include <hip/hip_bf16.h>
#include <stdint.h>

// Problem constants (match reference: B=128, S=1024, D=256, K=512)
#define N_PTS   (128 * 1024)
#define DDIM    256
#define KCLU    512
#define TILE_M  64          // rows of x per workgroup (4 waves x 16 rows)
#define LDS_STR 264         // 256 + 8 ushort padding -> kills LDS bank conflicts
#define PSPLIT  4           // partial buffers for the atomic-free scatter reduce

typedef __attribute__((ext_vector_type(16))) __bf16 v16bf;
typedef __attribute__((ext_vector_type(8)))  float  v8f;
typedef __attribute__((ext_vector_type(4)))  float  v4f;

union V16 { v16bf v; uint4 q[2]; };
union V8F { v8f  v; float f[8]; };

__device__ __forceinline__ unsigned short f2bf(float f) {
  union { float f; uint32_t u; } x; x.f = f;
  uint32_t r = (x.u + 0x7FFFu + ((x.u >> 16) & 1u)) >> 16;   // round-to-nearest-even
  return (unsigned short)r;
}

// ---------------------------------------------------------------------------
// Prep: centroids fp32 -> bf16 (L2-resident WMMA operand), and 0.5*||c||^2.
// One block per centroid row (256 threads).
// ---------------------------------------------------------------------------
__global__ void k_prep(const float* __restrict__ cent,
                       unsigned short* __restrict__ cbf,
                       float* __restrict__ c2h) {
  const int k = blockIdx.x;
  const int d = threadIdx.x;
  float c = cent[k * DDIM + d];
  cbf[k * DDIM + d] = f2bf(c);
  float s = c * c;
  #pragma unroll
  for (int m = 16; m >= 1; m >>= 1) s += __shfl_xor(s, m, 32);
  __shared__ float part[8];
  if ((threadIdx.x & 31) == 0) part[threadIdx.x >> 5] = s;
  __syncthreads();
  if (threadIdx.x == 0) {
    float t = 0.f;
    #pragma unroll
    for (int w = 0; w < 8; ++w) t += part[w];
    c2h[k] = 0.5f * t;
  }
}

// ---------------------------------------------------------------------------
// Pass 1: per-point argmin over K clusters via bf16 WMMA; stream the one-hot
// rows (nontemporal, never re-read) and write argmin indices for pass 2.
// Block = 128 threads = 4 waves; each wave owns 16 rows of the 64-row tile.
// ---------------------------------------------------------------------------
__global__ void __launch_bounds__(128)
k_assign(const float* __restrict__ x,
         const unsigned short* __restrict__ cbf,
         const float* __restrict__ c2h,
         float* __restrict__ onehot,
         int* __restrict__ idxArr) {
  __shared__ __align__(16) unsigned short xs[TILE_M * LDS_STR];

  const int t    = threadIdx.x;
  const int lid  = t & 31;
  const int wave = t >> 5;
  const int half = (lid >> 4) & 1;   // which 16-lane half of the wave32
  const int ln   = lid & 15;
  const size_t rowBase = (size_t)blockIdx.x * TILE_M;

  // Stage 64x256 fp32 tile -> bf16 in LDS (coalesced global b32 loads).
  for (int i = t; i < TILE_M * DDIM; i += 128) {
    const int r = i >> 8;       // /DDIM
    const int c = i & (DDIM - 1);
    xs[r * LDS_STR + c] = f2bf(x[(rowBase + r) * DDIM + c]);
  }
  __syncthreads();

  // A-fragment row for this lane (ISA 16-bit A layout: both halves carry M=0..15)
  const int mrow = wave * 16 + ln;

  float bestV[8];
  int   bestI[8];
  #pragma unroll
  for (int r = 0; r < 8; ++r) { bestV[r] = 3.4e38f; bestI[r] = 0; }

  for (int cc = 0; cc < KCLU / 16; ++cc) {
    const int   cIdx = cc * 16 + ln;         // this lane's cluster column
    const float c2v  = c2h[cIdx];

    V8F acc;
    #pragma unroll
    for (int r = 0; r < 8; ++r) acc.f[r] = 0.f;

    // B layout: half 0 -> K = {0..7, 16..23}, half 1 -> K = {8..15, 24..31}
    const unsigned short* brow = cbf + (size_t)cIdx * DDIM + half * 8;
    // Pull next chunk's B row toward the WGP while this chunk computes.
    __builtin_prefetch(brow + 16 * DDIM, 0, 1);   // global_prefetch_b8

    #pragma unroll
    for (int kb = 0; kb < DDIM / 32; ++kb) {
      // A layout: half 0 -> K = {0..7, 16..23}, half 1 -> K = {8..15, 24..31}
      const int kbase = kb * 32 + half * 8;
      V16 a, b;
      a.q[0] = *(const uint4*)(&xs[mrow * LDS_STR + kbase]);
      a.q[1] = *(const uint4*)(&xs[mrow * LDS_STR + kbase + 16]);
      const uint4* pb = (const uint4*)(brow + kb * 32);
      b.q[0] = pb[0];
      b.q[1] = pb[2];
      acc.v = __builtin_amdgcn_wmma_f32_16x16x32_bf16(
          /*neg_a=*/false, a.v, /*neg_b=*/false, b.v,
          /*c_mod=*/(short)0, acc.v, /*reuse_a=*/false, /*reuse_b=*/false);
    }

    // score = 0.5*||c||^2 - x.c  (x^2 row term irrelevant to argmin)
    #pragma unroll
    for (int r = 0; r < 8; ++r) {
      const float s = c2v - acc.f[r];
      if (s < bestV[r]) { bestV[r] = s; bestI[r] = cIdx; }  // strict < keeps first idx
    }
  }

  // Argmin butterfly across the 16 lanes of each half (C layout: M = r + 8*half)
  #pragma unroll
  for (int r = 0; r < 8; ++r) {
    #pragma unroll
    for (int m = 8; m >= 1; m >>= 1) {
      const float ov = __shfl_xor(bestV[r], m, 16);
      const int   oi = __shfl_xor(bestI[r], m, 16);
      if (ov < bestV[r] || (ov == bestV[r] && oi < bestI[r])) { bestV[r] = ov; bestI[r] = oi; }
    }
  }

  const size_t waveRow = rowBase + (size_t)wave * 16;

  // One lane per half records the argmin index of its 8 rows (pass-2 input).
  if (ln == 0) {
    #pragma unroll
    for (int r = 0; r < 8; ++r)
      idxArr[waveRow + half * 8 + r] = bestI[r];
  }

  // Stream the full one-hot rows in a single pass (nontemporal b128 stores;
  // nobody ever re-reads this 256 MB buffer, keep it out of the caches).
  for (int m = 0; m < 16; ++m) {
    const int idx = __shfl(bestI[m & 7], (m < 8) ? 0 : 16, 32);
    v4f* orow = (v4f*)(onehot + (waveRow + m) * KCLU);
    #pragma unroll 4
    for (int q = lid; q < KCLU / 4; q += 32) {
      const int c0 = q * 4;
      v4f v;
      v.x = (c0 + 0 == idx) ? 1.f : 0.f;
      v.y = (c0 + 1 == idx) ? 1.f : 0.f;
      v.z = (c0 + 2 == idx) ? 1.f : 0.f;
      v.w = (c0 + 3 == idx) ? 1.f : 0.f;
      __builtin_nontemporal_store(v, orow + q);
    }
  }
}

// ---------------------------------------------------------------------------
// Pass 2: atomic-free, deterministic scatter-reduce. One wave per
// (cluster, N-range): scan idx with b128 loads, ballot for matches,
// register-accumulate matching x rows (lane <-> d column, 8 VGPRs).
// Each x row is read by exactly one wave; partials land in PSPLIT disjoint
// buffers that are fully overwritten every call (no zeroing, no atomics).
// ---------------------------------------------------------------------------
__global__ void __launch_bounds__(32)
k_sums(const float* __restrict__ x,
       const int* __restrict__ idxArr,
       float* __restrict__ psums,
       float* __restrict__ pcounts) {
  const int k    = blockIdx.x & (KCLU - 1);
  const int p    = blockIdx.x >> 9;              // KCLU == 512
  const int lane = threadIdx.x;
  const int nspan = N_PTS / PSPLIT;              // 32768 points per range
  const int nbase = p * nspan;

  float acc[8];
  #pragma unroll
  for (int j = 0; j < 8; ++j) acc[j] = 0.f;
  int cnt = 0;

  const int4* ip = (const int4*)(idxArr + nbase);
  for (int it = 0; it < nspan / 128; ++it) {     // 256 scan iterations
    const int4 v = ip[it * 32 + lane];
    const int vv[4] = { v.x, v.y, v.z, v.w };
    #pragma unroll
    for (int c = 0; c < 4; ++c) {
      unsigned int bal = (unsigned int)__ballot(vv[c] == k);
      cnt += __popc(bal);
      while (bal) {
        const int l = __ffs(bal) - 1;
        bal &= bal - 1;
        const int n = nbase + it * 128 + l * 4 + c;
        const float* xr = x + (size_t)n * DDIM;
        #pragma unroll
        for (int j = 0; j < 8; ++j) acc[j] += xr[j * 32 + lane];
      }
    }
  }

  float* out = psums + (size_t)(p * KCLU + k) * DDIM;
  #pragma unroll
  for (int j = 0; j < 8; ++j) out[j * 32 + lane] = acc[j];
  if (lane == 0) pcounts[p * KCLU + k] = (float)cnt;
}

// Reduce the PSPLIT partials in fixed order, divide, EMA.
__global__ void k_final(const float* __restrict__ cent,
                        const float* __restrict__ psums,
                        const float* __restrict__ pcounts,
                        float* __restrict__ outc) {
  const int i = blockIdx.x * blockDim.x + threadIdx.x;   // K*D threads
  const int k = i >> 8;                                  // /DDIM
  float s = 0.f, c = 0.f;
  #pragma unroll
  for (int p = 0; p < PSPLIT; ++p) {
    s += psums[(size_t)p * KCLU * DDIM + i];
    c += pcounts[p * KCLU + k];
  }
  outc[i] = 0.9f * cent[i] + 0.1f * (s / c);
}

extern "C" void kernel_launch(void* const* d_in, const int* in_sizes, int n_in,
                              void* d_out, int out_size, void* d_ws, size_t ws_size,
                              hipStream_t stream) {
  const float* x    = (const float*)d_in[0];   // [N, D] fp32
  const float* cent = (const float*)d_in[1];   // [K, D] fp32
  float* onehot = (float*)d_out;                         // [N, K]
  float* outc   = onehot + (size_t)N_PTS * KCLU;         // [K, D]

  char* ws = (char*)d_ws;
  unsigned short* cbf = (unsigned short*)ws;             // K*D bf16      (262144 B)
  float* c2h     = (float*)(ws + 262144);                // K fp32        (  2048 B)
  int*   idxArr  = (int*)  (ws + 266240);                // N i32         (524288 B)
  float* psums   = (float*)(ws + 790528);                // P*K*D fp32    (2097152 B)
  float* pcounts = (float*)(ws + 2887680);               // P*K fp32      (  8192 B)

  k_prep  <<<KCLU, DDIM, 0, stream>>>(cent, cbf, c2h);
  k_assign<<<N_PTS / TILE_M, 128, 0, stream>>>(x, cbf, c2h, onehot, idxArr);
  k_sums  <<<KCLU * PSPLIT, 32, 0, stream>>>(x, idxArr, psums, pcounts);
  k_final <<<(KCLU * DDIM) / 256, 256, 0, stream>>>(cent, psums, pcounts, outc);
}
